// Joiner_82265803588315
// MI455X (gfx1250) — matrix operand order repaired and checked by reading, not compiled, not run on hardware
//
#include <hip/hip_runtime.h>

// ---------- types ----------
typedef __bf16 bf16_t;
typedef bf16_t v8bf  __attribute__((ext_vector_type(8)));
typedef bf16_t v16bf __attribute__((ext_vector_type(16)));
typedef float  v8f   __attribute__((ext_vector_type(8)));

// ---------- problem constants ----------
#define B_   8
#define T_   256
#define U_   64
#define E_   512
#define P_   640
#define J_   640
#define V_   1025
#define VP_  1040          // V padded to multiple of 16
#define LDST 648           // LDS row stride (bf16): 1296B = 324 dwords, 324%64=4 -> conflict-free rows

static __device__ __forceinline__ v16bf cat8(v8bf lo, v8bf hi) {
  return __builtin_shufflevector(lo, hi, 0,1,2,3,4,5,6,7,8,9,10,11,12,13,14,15);
}

// ---------- fp32 -> bf16 conversion ----------
__global__ void cvt_f32_bf16(const float* __restrict__ src, bf16_t* __restrict__ dst, int n) {
  int i = blockIdx.x * blockDim.x + threadIdx.x;
  if (i < n) dst[i] = (bf16_t)src[i];
}

// W_out (1025x640) -> bf16 padded to 1040x640, pad rows zeroed
__global__ void cvt_wout(const float* __restrict__ src, bf16_t* __restrict__ dst) {
  int i = blockIdx.x * blockDim.x + threadIdx.x;
  if (i < VP_ * J_) dst[i] = (i < V_ * J_) ? (bf16_t)src[i] : (bf16_t)0.0f;
}

// ---------- projection GEMM: out(rows x 640) = A(rows x K)[bf16] * W(640 x K)^T [bf16] + bias, fp32 out ----------
// One wave per 16x16 tile; K multiple of 32; rows multiple of 16. (~1% of total FLOPs)
__global__ void proj_gemm(const bf16_t* __restrict__ A, const bf16_t* __restrict__ W,
                          const float* __restrict__ bias, float* __restrict__ out, int K) {
  const int lane = threadIdx.x & 31;
  const int wave = threadIdx.x >> 5;
  const int nl   = lane & 15;     // N (and A-row) index within tile
  const int kg   = lane >> 4;     // lane-group selects K sub-range

  const int tile   = blockIdx.x * 8 + wave;
  const int m_tile = tile / (J_ / 16);
  const int n_tile = tile - m_tile * (J_ / 16);
  const int n      = n_tile * 16 + nl;

  const bf16_t* arow = A + (size_t)(m_tile * 16 + nl) * K + kg * 8;
  const bf16_t* wrow = W + (size_t)n * K + kg * 8;

  const float bv = bias[n];
  v8f c;
#pragma unroll
  for (int e = 0; e < 8; ++e) c[e] = bv;

#pragma unroll 4
  for (int k = 0; k < K; k += 32) {
    v8bf alo = *(const v8bf*)(arow + k);
    v8bf ahi = *(const v8bf*)(arow + k + 16);
    v8bf blo = *(const v8bf*)(wrow + k);
    v8bf bhi = *(const v8bf*)(wrow + k + 16);
    c = __builtin_amdgcn_wmma_f32_16x16x32_bf16(false, cat8(alo, ahi),
                                                false, cat8(blo, bhi),
                                                (short)0, c, false, false);
  }

  float* o0 = out + (size_t)(m_tile * 16 + kg * 8) * J_ + n_tile * 16 + nl;
#pragma unroll
  for (int i = 0; i < 8; ++i) o0[(size_t)i * J_] = c[i];
}

// ---------- fused joiner ----------
// One block per (b,t). Stage h = relu(enc_row + pred[u,:]) for all 64 u as bf16 in LDS,
// then each wave computes a 64x16 output strip: B (W_out tile) loaded ONCE per K-step,
// amortized over 4 WMMAs (4 independent accumulator chains).
__global__ void joiner_main(const float* __restrict__ encp,   // (B*T, 640) fp32
                            const float* __restrict__ predp,  // (B*U, 640) fp32
                            const bf16_t* __restrict__ Wout,  // (1040, 640) bf16
                            const float* __restrict__ bout,   // (1025) fp32
                            float* __restrict__ out) {        // (B,T,U,1025) fp32
  extern __shared__ char smem_raw[];
  bf16_t* h = (bf16_t*)smem_raw;    // 64 rows x 640, stride LDST

  const int bt  = blockIdx.x;       // b*T + t
  const int b   = bt >> 8;          // T_ = 256
  const int tid = threadIdx.x;

  const float* erow  = encp  + (size_t)bt * J_;
  const float* pbase = predp + (size_t)b * U_ * J_;

  // stage h: each thread owns columns {tid, tid+256, tid+512(<640)}
  const float e0 = erow[tid];
  const float e1 = erow[tid + 256];
  const float e2 = (tid < 128) ? erow[tid + 512] : 0.0f;
  for (int u = 0; u < U_; ++u) {
    const float* pr = pbase + (size_t)u * J_;
    bf16_t* hr = h + u * LDST;
    hr[tid]       = (bf16_t)fmaxf(e0 + pr[tid],       0.0f);
    hr[tid + 256] = (bf16_t)fmaxf(e1 + pr[tid + 256], 0.0f);
    if (tid < 128)
      hr[tid + 512] = (bf16_t)fmaxf(e2 + pr[tid + 512], 0.0f);
  }
  __syncthreads();

  const int lane = tid & 31;
  const int wave = tid >> 5;
  const int nl   = lane & 15;
  const int kg   = lane >> 4;

  // 65 N-tiles (v, padded to 1040) distributed over 8 waves; each strip covers all 64 u.
  for (int n_tile = wave; n_tile < VP_ / 16; n_tile += 8) {
    const int n = n_tile * 16 + nl;               // < 1040 (W_out padded)
    const float bv = (n < V_) ? bout[n] : 0.0f;

    v8f c0, c1, c2, c3;
#pragma unroll
    for (int e = 0; e < 8; ++e) { c0[e] = bv; c1[e] = bv; c2[e] = bv; c3[e] = bv; }

    const bf16_t* wrow = Wout + (size_t)n * J_ + kg * 8;               // L2-resident
    const bf16_t* a0 = h + (size_t)(0 * 16 + nl) * LDST + kg * 8;      // LDS, 16B aligned
    const bf16_t* a1 = a0 + 16 * LDST;
    const bf16_t* a2 = a0 + 32 * LDST;
    const bf16_t* a3 = a0 + 48 * LDST;

#pragma unroll 2
    for (int k = 0; k < J_; k += 32) {
      v8bf blo = *(const v8bf*)(wrow + k);
      v8bf bhi = *(const v8bf*)(wrow + k + 16);
      const v16bf bm = cat8(blo, bhi);

      c0 = __builtin_amdgcn_wmma_f32_16x16x32_bf16(false,
             cat8(*(const v8bf*)(a0 + k), *(const v8bf*)(a0 + k + 16)),
             false, bm, (short)0, c0, false, false);
      c1 = __builtin_amdgcn_wmma_f32_16x16x32_bf16(false,
             cat8(*(const v8bf*)(a1 + k), *(const v8bf*)(a1 + k + 16)),
             false, bm, (short)0, c1, false, false);
      c2 = __builtin_amdgcn_wmma_f32_16x16x32_bf16(false,
             cat8(*(const v8bf*)(a2 + k), *(const v8bf*)(a2 + k + 16)),
             false, bm, (short)0, c2, false, false);
      c3 = __builtin_amdgcn_wmma_f32_16x16x32_bf16(false,
             cat8(*(const v8bf*)(a3 + k), *(const v8bf*)(a3 + k + 16)),
             false, bm, (short)0, c3, false, false);
    }

    if (n < V_) {
      // C/D layout: VGPR i -> M = i (lanes 0-15) / i+8 (lanes 16-31), N = lane%16
      const int u0 = kg * 8;
      float* o = out + ((size_t)bt * U_ + u0) * V_ + n;
#pragma unroll
      for (int i = 0; i < 8; ++i) o[(size_t)i * V_] = c0[i];
      o += (size_t)16 * V_;
#pragma unroll
      for (int i = 0; i < 8; ++i) o[(size_t)i * V_] = c1[i];
      o += (size_t)16 * V_;
#pragma unroll
      for (int i = 0; i < 8; ++i) o[(size_t)i * V_] = c2[i];
      o += (size_t)16 * V_;
#pragma unroll
      for (int i = 0; i < 8; ++i) o[(size_t)i * V_] = c3[i];
    }
  }
}

// ---------- launch ----------
extern "C" void kernel_launch(void* const* d_in, const int* in_sizes, int n_in,
                              void* d_out, int out_size, void* d_ws, size_t ws_size,
                              hipStream_t stream) {
  (void)in_sizes; (void)n_in; (void)out_size; (void)ws_size;
  const float* enc    = (const float*)d_in[0];  // (8,256,512)
  const float* pred   = (const float*)d_in[1];  // (8,64,640)
  const float* W_enc  = (const float*)d_in[2];  // (640,512)
  const float* b_enc  = (const float*)d_in[3];  // (640)
  const float* W_pred = (const float*)d_in[4];  // (640,640)
  const float* b_pred = (const float*)d_in[5];  // (640)
  const float* W_out  = (const float*)d_in[6];  // (1025,640)
  const float* b_out  = (const float*)d_in[7];  // (1025)
  float* out = (float*)d_out;

  char* ws = (char*)d_ws;
  size_t off = 0;
  auto carve = [&](size_t bytes) { void* p = ws + off; off += (bytes + 255) & ~(size_t)255; return p; };

  bf16_t* enc_bf   = (bf16_t*)carve((size_t)B_ * T_ * E_ * 2);
  bf16_t* pred_bf  = (bf16_t*)carve((size_t)B_ * U_ * P_ * 2);
  bf16_t* wenc_bf  = (bf16_t*)carve((size_t)J_ * E_ * 2);
  bf16_t* wpred_bf = (bf16_t*)carve((size_t)J_ * P_ * 2);
  bf16_t* wout_bf  = (bf16_t*)carve((size_t)VP_ * J_ * 2);
  float*  encp     = (float*)carve((size_t)B_ * T_ * J_ * 4);
  float*  predp    = (float*)carve((size_t)B_ * U_ * J_ * 4);

  const int TPB = 256;
  auto blocks = [](int n, int t) { return (n + t - 1) / t; };

  // 1) bf16 conversions (L2-resident, negligible)
  cvt_f32_bf16<<<blocks(B_*T_*E_, TPB), TPB, 0, stream>>>(enc,    enc_bf,   B_*T_*E_);
  cvt_f32_bf16<<<blocks(B_*U_*P_, TPB), TPB, 0, stream>>>(pred,   pred_bf,  B_*U_*P_);
  cvt_f32_bf16<<<blocks(J_*E_,    TPB), TPB, 0, stream>>>(W_enc,  wenc_bf,  J_*E_);
  cvt_f32_bf16<<<blocks(J_*P_,    TPB), TPB, 0, stream>>>(W_pred, wpred_bf, J_*P_);
  cvt_wout    <<<blocks(VP_*J_,   TPB), TPB, 0, stream>>>(W_out,  wout_bf);

  // 2) projections (bf16 WMMA, fp32 out)
  {
    int tiles = (B_*T_/16) * (J_/16);   // 5120
    proj_gemm<<<tiles / 8, TPB, 0, stream>>>(enc_bf,  wenc_bf,  b_enc,  encp,  E_);
  }
  {
    int tiles = (B_*U_/16) * (J_/16);   // 1280
    proj_gemm<<<tiles / 8, TPB, 0, stream>>>(pred_bf, wpred_bf, b_pred, predp, P_);
  }

  // 3) fused broadcast-add + ReLU + output GEMM, one block per (b,t)
  const size_t lds_bytes = (size_t)U_ * LDST * sizeof(bf16_t);  // 82,944 B
  joiner_main<<<B_ * T_, TPB, lds_bytes, stream>>>(encp, predp, wout_bf, b_out, out);
}
